// Transformer_19756849562056
// MI455X (gfx1250) — matrix operand order repaired and checked by reading, not compiled, not run on hardware
//
#include <hip/hip_runtime.h>

#define B_  32
#define N_  3072
#define C_  128
#define L_  4
#define K_  6
#define NB_ 8
#define TM_ 512
#define CM_ 512
#define KC_ (K_ * C_)   // 768: concatenated Chebyshev K dimension

typedef __attribute__((ext_vector_type(16))) __bf16 v16bf;
typedef __attribute__((ext_vector_type(8)))  float  v8f;
typedef __attribute__((ext_vector_type(4)))  float  f4;
typedef __attribute__((ext_vector_type(4)))  __bf16 bf4;

union Frag { v16bf v; f4 q[2]; };

__device__ __forceinline__ float gelu_f(float x) {
  float x3 = x * x * x;
  return 0.5f * x * (1.f + tanhf(0.79788456080286535588f * (x + 0.044715f * x3)));
}

__device__ __forceinline__ void load_a(Frag& a, const __bf16* __restrict__ ap, int k0) {
  a.q[0] = *(const f4*)(ap + k0);
  a.q[1] = *(const f4*)(ap + k0 + 8);
}

__device__ __forceinline__ void load_b4(Frag* b, const __bf16* __restrict__ bp,
                                        int K, int k0) {
#pragma unroll
  for (int t = 0; t < 4; ++t) {
    const __bf16* bk = bp + (long)t * 16 * K + k0;
    b[t].q[0] = *(const f4*)(bk);
    b[t].q[1] = *(const f4*)(bk + 8);
  }
}

__device__ __forceinline__ void wmma4(v8f* acc, const Frag& a, const Frag* b) {
#pragma unroll
  for (int t = 0; t < 4; ++t)
    acc[t] = __builtin_amdgcn_wmma_f32_16x16x32_bf16(
        false, a.v, false, b[t].v, (short)0, acc[t], false, false);
}

// ---------------------------------------------------------------------------
// bf16 WMMA GEMM:  C[M,N] = A[M,K](bf16,row-major) * Bt[N,K]^T (+bias)(+gelu)
// 128 threads = 4 waves; wave w computes rows [m0,m0+16) over a 64-wide column
// strip (4 v8f accumulators; A fragment reused 4x per K-step).
// Explicit ping-pong fragment double buffering: while the 4 WMMAs of one
// buffer execute, the other buffer's 10 b128 loads are in flight, so waits
// are partial instead of loadcnt==0 stalls. Requires K % 64 == 0 (true for
// all shapes here: 768 / 3072 / 512 / 128).
// O16: epilogue emits bf16 (feeds the next GEMM directly, no f32 round trip).
// ACC: epilogue accumulates into existing f32 C (residual add).
// ---------------------------------------------------------------------------
template <bool BIAS, bool GELU, bool ACC, bool O16>
__global__ __launch_bounds__(128)
void gemm_bf16_wmma(const __bf16* __restrict__ A, long sA,
                    const __bf16* __restrict__ Bt,
                    const float* __restrict__ bias,
                    void* __restrict__ Cp, long sC,
                    int M, int N, int K) {
  const int lane = threadIdx.x & 31;
  const int wv   = threadIdx.x >> 5;
  const int m0   = (blockIdx.y * 4 + wv) * 16;
  const int n0   = blockIdx.x * 64;
  if (m0 >= M) return;
  A += (long)blockIdx.z * sA;

  const int ml = lane & 15;   // A row / B column / C column within tile
  const int h  = lane >> 4;   // half-wave selects K sub-block

  // A fragment: lane holds row (m0+ml); elements e at K=e+16h
  // -> two contiguous 16B loads.
  const __bf16* ap = A + (long)(m0 + ml) * K + 16 * h;
  // B fragment from Bt (N x K row-major): lane holds column (n0+t*16+ml);
  // elements e at K=e+16h -> two contiguous 16B loads.
  const __bf16* bp = Bt + (long)(n0 + ml) * K + 16 * h;

  const v8f vzero = {0.f, 0.f, 0.f, 0.f, 0.f, 0.f, 0.f, 0.f};
  v8f acc[4];
#pragma unroll
  for (int t = 0; t < 4; ++t) acc[t] = vzero;

  // Prologue: fill both buffers (k = 0 and k = 32).
  Frag a0, a1, b0[4], b1[4];
  load_a(a0, ap, 0);
  load_b4(b0, bp, K, 0);
  load_a(a1, ap, 32);
  load_b4(b1, bp, K, 32);

  int k0 = 0;
  for (; k0 + 64 < K; k0 += 64) {
    wmma4(acc, a0, b0);
    load_a(a0, ap, k0 + 64);
    load_b4(b0, bp, K, k0 + 64);
    __builtin_prefetch(ap + k0 + 128, 0, 3);
    wmma4(acc, a1, b1);
    load_a(a1, ap, k0 + 96);
    load_b4(b1, bp, K, k0 + 96);
  }
  wmma4(acc, a0, b0);
  wmma4(acc, a1, b1);

  // Epilogue. C/D layout: lane -> column n = lane&15; VGPR r -> row r + 8h.
#pragma unroll
  for (int t = 0; t < 4; ++t) {
    const int n = n0 + t * 16 + ml;
    float bv = 0.f;
    if (BIAS) bv = bias[n];
#pragma unroll
    for (int r = 0; r < 8; ++r) {
      const int m = m0 + r + 8 * h;
      float v = acc[t][r] + bv;
      if (GELU) v = gelu_f(v);
      if (O16) {
        __bf16* cp = (__bf16*)Cp + (long)blockIdx.z * sC + (long)m * N + n;
        *cp = (__bf16)v;
      } else {
        float* cp = (float*)Cp + (long)blockIdx.z * sC + (long)m * N + n;
        if (ACC) v += *cp;
        *cp = v;
      }
    }
  }
}

// ---------------------------------------------------------------------------
// LayerNorm over C=128 channels. One wave32 per row, 4 channels per lane,
// wave reductions via shfl_xor. Optional fused input GELU; writes f32 and/or
// bf16 (bf16 row stride ld16 so T_0 can land inside the concat buffer).
// ---------------------------------------------------------------------------
template <bool GIN, bool W32, bool W16>
__global__ __launch_bounds__(128)
void ln_k(const float* __restrict__ in, const float* __restrict__ sc,
          const float* __restrict__ bi, float* __restrict__ o32,
          __bf16* __restrict__ o16, long ld16, int rows) {
  const int row = blockIdx.x * 4 + (threadIdx.x >> 5);
  if (row >= rows) return;
  const int lane = threadIdx.x & 31;
  f4 v = *(const f4*)(in + (long)row * C_ + lane * 4);
  if (GIN) { v.x = gelu_f(v.x); v.y = gelu_f(v.y); v.z = gelu_f(v.z); v.w = gelu_f(v.w); }
  float s = v.x + v.y + v.z + v.w;
#pragma unroll
  for (int off = 16; off > 0; off >>= 1) s += __shfl_xor(s, off, 32);
  const float mean = s * (1.f / C_);
  f4 d; d.x = v.x - mean; d.y = v.y - mean; d.z = v.z - mean; d.w = v.w - mean;
  float q = d.x * d.x + d.y * d.y + d.z * d.z + d.w * d.w;
#pragma unroll
  for (int off = 16; off > 0; off >>= 1) q += __shfl_xor(q, off, 32);
  const float r = rsqrtf(q * (1.f / C_) + 1e-6f);
  f4 scv = *(const f4*)(sc + lane * 4);
  f4 biv = *(const f4*)(bi + lane * 4);
  f4 o;
  o.x = d.x * r * scv.x + biv.x;
  o.y = d.y * r * scv.y + biv.y;
  o.z = d.z * r * scv.z + biv.z;
  o.w = d.w * r * scv.w + biv.w;
  if (W32) *(f4*)(o32 + (long)row * C_ + lane * 4) = o;
  if (W16) {
    bf4 ob;
    ob.x = (__bf16)o.x; ob.y = (__bf16)o.y; ob.z = (__bf16)o.z; ob.w = (__bf16)o.w;
    *(bf4*)(o16 + (long)row * ld16 + lane * 4) = ob;
  }
}

// ---------------------------------------------------------------------------
// Chebyshev recurrence step: out = alpha*lap(in) + beta*other, where
// lap(in)[b,n,c] = sum_k nbr_w[n,k] * in[b, nbr_idx[n,k], c].
// Writes f32 state plus bf16 copy at column slot k of the concat buffer.
// `other` may alias `o32` (each thread touches only its own element).
// ---------------------------------------------------------------------------
template <bool HASO>
__global__ __launch_bounds__(128)
void lap_k(const float* __restrict__ in, const float* __restrict__ oth,
           const int* __restrict__ idx, const float* __restrict__ w,
           float alpha, float beta,
           float* __restrict__ o32, __bf16* __restrict__ o16, long ld16) {
  const int n = blockIdx.x, b = blockIdx.y, c = threadIdx.x;
  float acc = 0.f;
#pragma unroll
  for (int k = 0; k < NB_; ++k) {
    const int j = idx[n * NB_ + k];
    acc += w[n * NB_ + k] * in[((long)b * N_ + j) * C_ + c];
  }
  float v = alpha * acc;
  const long row = (long)b * N_ + n;
  if (HASO) v += beta * oth[row * C_ + c];
  o32[row * C_ + c] = v;
  o16[row * ld16 + c] = (__bf16)v;
}

// ---------------------------------------------------------------------------
// Batched transpose + bf16 cast: src[z] is R x Cc (f32) -> dst[z] is Cc x R.
// ---------------------------------------------------------------------------
__global__ __launch_bounds__(256)
void tcast_k(const float* __restrict__ src, __bf16* __restrict__ dst,
             int R, int Cc) {
  __shared__ float tile[32][33];
  const long sl = (long)R * Cc;
  src += (long)blockIdx.z * sl;
  dst += (long)blockIdx.z * sl;
  const int r0 = blockIdx.y * 32, c0 = blockIdx.x * 32;
  const int tx = threadIdx.x & 31, ty = threadIdx.x >> 5;
#pragma unroll
  for (int i = ty; i < 32; i += 8)
    tile[i][tx] = src[(long)(r0 + i) * Cc + c0 + tx];
  __syncthreads();
#pragma unroll
  for (int i = ty; i < 32; i += 8)
    dst[(long)(c0 + i) * R + r0 + tx] = (__bf16)tile[tx][i];
}

// Pack conv weights: dst[li][f][k*C+c] = src[li][k][c][f]  (bf16, Bt layout
// for the fused K=768 Chebyshev GEMM).
__global__ __launch_bounds__(256)
void wpack_conv_k(const float* __restrict__ src, __bf16* __restrict__ dst) {
  const int k  = blockIdx.x;   // 0..K_-1
  const int li = blockIdx.y;   // 0..L_*2-1
  const float* s = src + ((long)li * K_ + k) * C_ * C_;
  __bf16* d = dst + (long)li * C_ * KC_ + (long)k * C_;
  for (int idx = threadIdx.x; idx < C_ * C_; idx += 256) {
    const int c = idx >> 7, f = idx & 127;
    d[(long)f * KC_ + c] = (__bf16)s[(long)c * C_ + f];
  }
}

// x[b,n,c] += g[b,c,n]   (token-mixing output transposed back into residual)
__global__ __launch_bounds__(256)
void addtr_k(float* __restrict__ x, const float* __restrict__ g) {
  __shared__ float tile[32][33];
  const int b = blockIdx.z;
  const int n0 = blockIdx.y * 32, c0 = blockIdx.x * 32;
  const float* gp = g + (long)b * C_ * N_;
  float* xp = x + (long)b * N_ * C_;
  const int tx = threadIdx.x & 31, ty = threadIdx.x >> 5;
#pragma unroll
  for (int i = ty; i < 32; i += 8)
    tile[i][tx] = gp[(long)(c0 + i) * N_ + n0 + tx];
  __syncthreads();
#pragma unroll
  for (int i = ty; i < 32; i += 8)
    xp[(long)(n0 + i) * C_ + c0 + tx] += tile[tx][i];
}

// ---------------------------------------------------------------------------
extern "C" void kernel_launch(void* const* d_in, const int* in_sizes, int n_in,
                              void* d_out, int out_size, void* d_ws, size_t ws_size,
                              hipStream_t stream) {
  const float* maps   = (const float*)d_in[0];
  const float* nbr_w  = (const float*)d_in[1];
  const float* ln_s   = (const float*)d_in[2];
  const float* ln_b   = (const float*)d_in[3];
  const float* conv_W = (const float*)d_in[4];
  const float* conv_b = (const float*)d_in[5];
  const float* tok_W1 = (const float*)d_in[6];
  const float* tok_b1 = (const float*)d_in[7];
  const float* tok_W2 = (const float*)d_in[8];
  const float* tok_b2 = (const float*)d_in[9];
  const float* ch_W1  = (const float*)d_in[10];
  const float* ch_b1  = (const float*)d_in[11];
  const float* ch_W2  = (const float*)d_in[12];
  const float* ch_b2  = (const float*)d_in[13];
  const int*   nbr_i  = (const int*)d_in[14];
  (void)in_sizes; (void)n_in; (void)out_size; (void)ws_size;

  char* wsp = (char*)d_ws;
  size_t off = 0;
  auto alloc = [&](size_t bytes) -> char* {
    char* p = wsp + off;
    off += (bytes + 255) & ~(size_t)255;
    return p;
  };
  const long BNC  = (long)B_ * N_ * C_;
  const int  rows = B_ * N_;
  float*  y32   = (float*)alloc(BNC * 4);                    // LN out / T state
  float*  t2    = (float*)alloc(BNC * 4);                    // T state
  float*  cvout = (float*)alloc(BNC * 4);                    // conv out / g2
  __bf16* tcat  = (__bf16*)alloc((long)rows * KC_ * 2);      // [T0..T5] / MLP hidden bf16
  __bf16* abf   = (__bf16*)alloc(BNC * 2);                   // bf16 A staging
  __bf16* wcv   = (__bf16*)alloc((long)L_ * 2 * C_ * KC_ * 2);
  __bf16* wt1   = (__bf16*)alloc((long)L_ * TM_ * N_ * 2);
  __bf16* wt2   = (__bf16*)alloc((long)L_ * N_ * TM_ * 2);
  __bf16* wc1   = (__bf16*)alloc((long)L_ * CM_ * C_ * 2);
  __bf16* wc2   = (__bf16*)alloc((long)L_ * C_ * CM_ * 2);

  float* x = (float*)d_out;
  hipMemcpyAsync(x, maps, BNC * 4, hipMemcpyDeviceToDevice, stream);

  // Pre-pack all weights to N x K bf16 once (L2-resident for the whole run).
  wpack_conv_k<<<dim3(K_, L_ * 2), 256, 0, stream>>>(conv_W, wcv);
  tcast_k<<<dim3(TM_ / 32, N_ / 32, L_), 256, 0, stream>>>(tok_W1, wt1, N_, TM_);
  tcast_k<<<dim3(N_ / 32, TM_ / 32, L_), 256, 0, stream>>>(tok_W2, wt2, TM_, N_);
  tcast_k<<<dim3(CM_ / 32, C_ / 32, L_), 256, 0, stream>>>(ch_W1, wc1, C_, CM_);
  tcast_k<<<dim3(C_ / 32, CM_ / 32, L_), 256, 0, stream>>>(ch_W2, wc2, CM_, C_);

  const dim3 lapg(N_, B_);

  // Chebyshev conv: build [T0|...|T5] (M x 768 bf16), then ONE K=768 GEMM:
  // cvout = Tcat @ Wcat + bconv.  (No accumulate round trips through HBM.)
  auto run_conv = [&](int li) {
    lap_k<false><<<lapg, 128, 0, stream>>>(y32, nullptr, nbr_i, nbr_w,
                                           1.f, 0.f, t2, tcat + 1 * C_, KC_);
    float* prev = y32;
    float* cur  = t2;
    for (int k = 2; k < K_; ++k) {
      lap_k<true><<<lapg, 128, 0, stream>>>(cur, prev, nbr_i, nbr_w,
                                            2.f, -1.f, prev, tcat + (long)k * C_, KC_);
      float* t = prev; prev = cur; cur = t;
    }
    gemm_bf16_wmma<true, false, false, false><<<dim3(C_ / 64, rows / 64, 1), 128, 0, stream>>>(
        tcat, 0, wcv + (long)li * C_ * KC_, conv_b + (long)li * C_,
        cvout, 0, rows, C_, KC_);
  };

  for (int l = 0; l < L_; ++l) {
    const float* s0 = ln_s + ((long)l * 4 + 0) * C_; const float* b0 = ln_b + ((long)l * 4 + 0) * C_;
    const float* s1 = ln_s + ((long)l * 4 + 1) * C_; const float* b1 = ln_b + ((long)l * 4 + 1) * C_;
    const float* s2 = ln_s + ((long)l * 4 + 2) * C_; const float* b2 = ln_b + ((long)l * 4 + 2) * C_;
    const float* s3 = ln_s + ((long)l * 4 + 3) * C_; const float* b3 = ln_b + ((long)l * 4 + 3) * C_;

    // ---- half 1: Chebyshev conv + token-mixing MLP ----
    ln_k<false, true, true><<<rows / 4, 128, 0, stream>>>(x, s0, b0, y32, tcat, KC_, rows);
    run_conv(l * 2 + 0);
    ln_k<true, true, false><<<rows / 4, 128, 0, stream>>>(cvout, s1, b1, y32, nullptr, 0, rows);
    // y^T per batch: (N,C) -> (C,N) bf16
    tcast_k<<<dim3(C_ / 32, N_ / 32, B_), 256, 0, stream>>>(y32, abf, N_, C_);
    // h1 = gelu(yT @ W1 + b1) -> bf16 directly (no f32 round trip)
    gemm_bf16_wmma<true, true, false, true><<<dim3(TM_ / 64, C_ / 64, B_), 128, 0, stream>>>(
        abf, (long)C_ * N_, wt1 + (long)l * TM_ * N_, tok_b1 + (long)l * TM_,
        tcat, (long)C_ * TM_, C_, TM_, N_);
    gemm_bf16_wmma<true, false, false, false><<<dim3(N_ / 64, C_ / 64, B_), 128, 0, stream>>>(
        tcat, (long)C_ * TM_, wt2 + (long)l * N_ * TM_, tok_b2 + (long)l * N_,
        cvout, (long)C_ * N_, C_, N_, TM_);
    addtr_k<<<dim3(C_ / 32, N_ / 32, B_), 256, 0, stream>>>(x, cvout);

    // ---- half 2: Chebyshev conv + channel MLP ----
    ln_k<false, true, true><<<rows / 4, 128, 0, stream>>>(x, s2, b2, y32, tcat, KC_, rows);
    run_conv(l * 2 + 1);
    ln_k<true, false, true><<<rows / 4, 128, 0, stream>>>(cvout, s3, b3, nullptr, abf, C_, rows);
    gemm_bf16_wmma<true, true, false, true><<<dim3(CM_ / 64, rows / 64, 1), 128, 0, stream>>>(
        abf, 0, wc1 + (long)l * CM_ * C_, ch_b1 + (long)l * CM_,
        tcat, 0, rows, CM_, C_);
    gemm_bf16_wmma<true, false, true, false><<<dim3(C_ / 64, rows / 64, 1), 128, 0, stream>>>(
        tcat, 0, wc2 + (long)l * C_ * CM_, ch_b2 + (long)l * C_,
        x, 0, rows, C_, CM_);
  }
}
